// YOLOV8LabelEncoder_65025804861655
// MI455X (gfx1250) — matrix operand order repaired and checked by reading, not compiled, not run on hardware
//
#include <hip/hip_runtime.h>
#include <hip/hip_bf16.h>

typedef __attribute__((ext_vector_type(16))) _Float16 v16h;
typedef __attribute__((ext_vector_type(8)))  float    v8f;

#define MAX_A   8448   // LDS cap for the am array (A = 8400)
#define MAXK    10
#define CEPS    1e-7f  // keras epsilon inside compute_ciou
#define NEPS    1e-9f  // EPSILON for norm_am denominator

// ---------------------------------------------------------------------------
// CIoU(gt=b1, decode=b2); atan(w1/h1) for the GT box is hoisted (a1).
// ---------------------------------------------------------------------------
__device__ __forceinline__ float ciou_f(float gx1, float gy1, float gx2, float gy2,
                                        float w1, float h1, float a1,
                                        float bx1, float by1, float bx2, float by2) {
    float w2 = bx2 - bx1, h2 = by2 - by1 + CEPS;
    float iw = fminf(gx2, bx2) - fmaxf(gx1, bx1);
    float ih = fminf(gy2, by2) - fmaxf(gy1, by1);
    float inter = fmaxf(iw, 0.f) * fmaxf(ih, 0.f);
    float uni = w1 * h1 + w2 * h2 - inter + CEPS;
    float iou = inter / uni;
    float cw = fmaxf(gx2, bx2) - fminf(gx1, bx1);
    float ch = fmaxf(gy2, by2) - fminf(gy1, by1);
    float c2 = cw * cw + ch * ch + CEPS;
    float dx = bx1 + bx2 - gx1 - gx2;
    float dy = by1 + by2 - gy1 - gy2;
    float rho2 = (dx * dx + dy * dy) * 0.25f;
    float da = atanf(w2 / h2) - a1;
    float v = (4.0f / (float)(M_PI * M_PI)) * da * da;
    float alpha = v / (v - iou + (1.0f + CEPS));
    return iou - (rho2 / c2 + v * alpha);
}

// ---------------------------------------------------------------------------
// Kernel 0: zero the atomic-scatter workspace.
// ---------------------------------------------------------------------------
__global__ void init_ws(unsigned long long* __restrict__ pack,
                        unsigned* __restrict__ normb, int n) {
    int i = blockIdx.x * 256 + threadIdx.x;
    if (i < n) { pack[i] = 0ull; normb[i] = 0u; }
}

// ---------------------------------------------------------------------------
// Kernel W (WMMA path): bbox_scores[b,g,a] = scores[b,a,label_g] computed as
// P = onehot(labels)(GxC) x scores^T(CxA) with v_wmma_f32_16x16x32_f16.
// One wave per 16g x 16a tile; 3 k-steps cover C=80 (pad to 96 with zeros).
// ---------------------------------------------------------------------------
__global__ __launch_bounds__(256) void gather_scores_wmma(
        const float* __restrict__ scores, const int* __restrict__ gt_labels,
        _Float16* __restrict__ out, int A, int C, int G) {
    const int lane = threadIdx.x & 31;
    const int wave = threadIdx.x >> 5;                 // 0..7 (wave32)
    const int half = lane >> 4;                        // lane group
    const int l16  = lane & 15;
    const int abase = blockIdx.x * 128 + wave * 16;
    const int gbase = blockIdx.y * 16;
    const int b     = blockIdx.z;

    const int gl  = gbase + l16;
    const int lab = (gl < G) ? gt_labels[(size_t)b * G + gl] : -1;  // one-hot row

    const int a = abase + l16;
    const float* srow = scores + ((size_t)b * A + (size_t)(a < A ? a : 0)) * C;

    v8f acc = {0.f, 0.f, 0.f, 0.f, 0.f, 0.f, 0.f, 0.f};
#pragma unroll
    for (int kt = 0; kt < 3; ++kt) {
        const int kbase = kt * 32;
        const int r = lab - kbase;                     // relative one-hot column
        v16h av, bv;
        // A fragment (16x32 f16, contiguous layout: K = e + (half?16:0))
#pragma unroll
        for (int e = 0; e < 16; ++e) {
            int K = e + (half ? 16 : 0);
            av[e] = (r == K) ? (_Float16)1.0f : (_Float16)0.0f;
        }
        // B fragment (32x16): lane column a, halves split K 0..15 / 16..31
        const int cstart = kbase + half * 16;
        if (a < A && cstart < C) {
            const float4* p = (const float4*)(srow + cstart);  // 16B aligned (C=80)
#pragma unroll
            for (int q = 0; q < 4; ++q) {
                float4 f = p[q];
                bv[4 * q + 0] = (_Float16)f.x; bv[4 * q + 1] = (_Float16)f.y;
                bv[4 * q + 2] = (_Float16)f.z; bv[4 * q + 3] = (_Float16)f.w;
            }
        } else {
#pragma unroll
            for (int e = 0; e < 16; ++e) bv[e] = (_Float16)0.0f;
        }
        acc = __builtin_amdgcn_wmma_f32_16x16x32_f16(
                  false, av, false, bv, (short)0, acc, false, false);
    }
    // D layout: lanes 0-15 VGPR j -> M=j; lanes 16-31 -> M=8+j; N = l16
#pragma unroll
    for (int j = 0; j < 8; ++j) {
        int g = gbase + half * 8 + j;
        if (g < G && a < A)
            out[((size_t)b * G + g) * A + a] = (_Float16)acc[j];
    }
}

// ---------------------------------------------------------------------------
// Kernel 1: one block per (b,g). am -> LDS, top-10 by repeated block argmax,
// scatter argmax-over-g (packed u64) and norm_am via atomicMax.
// ---------------------------------------------------------------------------
__global__ __launch_bounds__(256) void yolo_match(
        const float* __restrict__ scores, const float* __restrict__ decode,
        const float* __restrict__ anchors, const int* __restrict__ gt_labels,
        const float* __restrict__ gt_bboxes, const unsigned char* __restrict__ gt_mask,
        const _Float16* __restrict__ gathered, int useGathered,
        unsigned long long* __restrict__ pack, unsigned* __restrict__ normb,
        int A, int C, int G) {
    __shared__ float s_am[MAX_A];
    __shared__ float rv[256];
    __shared__ int   ri[256];
    __shared__ int   s_cidx[MAXK];
    __shared__ float s_cam[MAXK];
    __shared__ float s_cov[MAXK];
    __shared__ float s_maxov;
    __shared__ int   s_n, s_stop;

    const int t   = threadIdx.x;
    const int blk = blockIdx.x;            // b*G + g
    const int b   = blk / G;
    const int g   = blk % G;
    if (!gt_mask[blk]) return;             // uniform: whole block exits

    const float gx1 = gt_bboxes[(size_t)blk * 4 + 0];
    const float gy1 = gt_bboxes[(size_t)blk * 4 + 1];
    const float gx2 = gt_bboxes[(size_t)blk * 4 + 2];
    const float gy2 = gt_bboxes[(size_t)blk * 4 + 3];
    const int   lab = gt_labels[blk];
    const float w1 = gx2 - gx1, h1 = gy2 - gy1 + CEPS;
    const float a1 = atanf(w1 / h1);

    const float*     dec_b = decode + (size_t)b * A * 4;
    const float*     sc_b  = scores + (size_t)b * A * C;
    const _Float16*  gw    = gathered + ((size_t)b * G + g) * A;

    // Phase 1: am into LDS
    for (int a = t; a < A; a += 256) {
        float ax = anchors[2 * a], ay = anchors[2 * a + 1];
        float am = 0.f;
        bool inb = (gx1 < ax) && (gy1 < ay) && (gx2 > ax) && (gy2 > ay);
        if (inb) {
            float4 bx = ((const float4*)dec_b)[a];
            float sc = useGathered ? fmaxf((float)gw[a], 0.f)
                                   : sc_b[(size_t)a * C + lab];
            float ov = ciou_f(gx1, gy1, gx2, gy2, w1, h1, a1, bx.x, bx.y, bx.z, bx.w);
            float o2 = ov * ov;
            am = sqrtf(sc) * (o2 * o2 * o2);           // score^0.5 * ciou^6
        }
        s_am[a] = am;
    }
    if (t == 0) { s_n = 0; s_stop = 0; }
    __syncthreads();

    // Phase 2: top-10 (metric > 0), smallest-index tie-break (top_k stability)
    for (int k = 0; k < MAXK; ++k) {
        float bv = 0.f; int bi = -1;
        for (int a = t; a < A; a += 256) {
            float v = s_am[a];
            if (v > bv) { bv = v; bi = a; }            // ascending a => first max
        }
        rv[t] = bv; ri[t] = bi;
        __syncthreads();
        for (int s = 128; s; s >>= 1) {
            if (t < s) {
                float ov2 = rv[t + s]; int oi = ri[t + s];
                if (ov2 > rv[t] ||
                    (ov2 == rv[t] && oi != -1 && (ri[t] == -1 || oi < ri[t]))) {
                    rv[t] = ov2; ri[t] = oi;
                }
            }
            __syncthreads();
        }
        if (t == 0) {
            if (ri[0] >= 0 && rv[0] > 0.f) {
                s_cidx[k] = ri[0]; s_cam[k] = rv[0];
                s_am[ri[0]] = -1.f;                    // exclude from next rounds
                s_n = k + 1;
            } else s_stop = 1;
        }
        __syncthreads();
        if (s_stop) break;
    }
    const int n = s_n;
    if (n == 0) return;

    // Phase 3: recompute CIoU only for the <=10 matched anchors
    if (t < n) {
        int a = s_cidx[t];
        float4 bx = ((const float4*)dec_b)[a];
        s_cov[t] = ciou_f(gx1, gy1, gx2, gy2, w1, h1, a1, bx.x, bx.y, bx.z, bx.w);
    }
    __syncthreads();
    if (t == 0) {
        float m = 0.f;                                  // max includes zeros
        for (int k = 0; k < n; ++k) m = fmaxf(m, s_cov[k]);
        s_maxov = m;
    }
    __syncthreads();

    // Phase 4: scatter. norm contribution >= 0 so float bits are max-ordered.
    if (t < n) {
        int a = s_cidx[t];
        float contrib = s_cam[t] * s_maxov / (s_cam[0] + NEPS);
        atomicMax(&normb[(size_t)b * A + a], __float_as_uint(contrib));
        float ovk = s_cov[t];
        if (ovk > 0.f) {                                // only >0 affects gmask/argmax
            unsigned long long pk =
                (((unsigned long long)__float_as_uint(ovk)) << 32) |
                (unsigned long long)(0xFFFFFFFFu - (unsigned)g);  // ties -> smaller g
            atomicMax(&pack[(size_t)b * A + a], pk);
        }
    }
}

// ---------------------------------------------------------------------------
// Kernel 2a: per (b,a) bbox (float4, coalesced), fg = 1.0 (argmax >= 0 always),
// and class id (-1 if unmatched) into workspace.
// ---------------------------------------------------------------------------
__global__ void finalize_a(const unsigned long long* __restrict__ pack,
                           const float* __restrict__ gt_bboxes,
                           const int* __restrict__ gt_labels,
                           float* __restrict__ out_bbox, float* __restrict__ out_fg,
                           int* __restrict__ clsw, int A, int G, int total) {
    int i = blockIdx.x * 256 + threadIdx.x;
    if (i >= total) return;
    int b = i / A;
    unsigned long long p = pack[i];
    unsigned ovb = (unsigned)(p >> 32);
    float4 bb; int cls = -1;
    if (ovb != 0u) {
        int g = (int)(0xFFFFFFFFu - (unsigned)(p & 0xFFFFFFFFull));
        const float* gp = gt_bboxes + ((size_t)b * G + g) * 4;
        bb = make_float4(gp[0], gp[1], gp[2], gp[3]);
        cls = gt_labels[(size_t)b * G + g];
    } else {
        bb = make_float4(-1.f, -1.f, -1.f, -1.f);
    }
    ((float4*)out_bbox)[i] = bb;
    out_fg[i] = 1.0f;
    clsw[i]   = cls;
}

// ---------------------------------------------------------------------------
// Kernel 2b: fully coalesced class one-hot * norm_am over B*A*C.
// ---------------------------------------------------------------------------
__global__ void finalize_b(const int* __restrict__ clsw,
                           const unsigned* __restrict__ normb,
                           float* __restrict__ out_cls, int C, long long total) {
    long long i = (long long)blockIdx.x * 256 + threadIdx.x;
    if (i >= total) return;
    int c = (int)(i % C);
    long long ba = i / C;
    int cl = clsw[ba];
    out_cls[i] = (cl == c) ? __uint_as_float(normb[ba]) : 0.f;
}

// ---------------------------------------------------------------------------
extern "C" void kernel_launch(void* const* d_in, const int* in_sizes, int n_in,
                              void* d_out, int out_size, void* d_ws, size_t ws_size,
                              hipStream_t stream) {
    const float* scores     = (const float*)d_in[0];
    const float* decode     = (const float*)d_in[1];
    const float* anchors    = (const float*)d_in[2];
    const int*   gt_labels  = (const int*)d_in[3];
    const float* gt_bboxes  = (const float*)d_in[4];
    const unsigned char* gt_mask = (const unsigned char*)d_in[5];

    const int A = in_sizes[2] / 2;                               // anchors (A,2)
    const int B = (int)((long long)in_sizes[1] / ((long long)A * 4)); // decode (B,A,4)
    const int C = (int)((long long)in_sizes[0] / ((long long)B * A)); // scores (B,A,C)
    const int G = in_sizes[3] / B;                               // labels (B,G)
    const size_t nBA = (size_t)B * A;

    // Workspace layout
    char* ws = (char*)d_ws;
    unsigned long long* pack  = (unsigned long long*)ws;          // 8B * nBA
    unsigned*           normb = (unsigned*)(ws + nBA * 8);        // 4B * nBA
    int*                clsw  = (int*)(ws + nBA * 12);            // 4B * nBA
    size_t g16off = (nBA * 16 + 255) & ~(size_t)255;
    _Float16* g16 = (_Float16*)(ws + g16off);                     // 2B * B*G*A (optional)
    size_t needWmma = g16off + (size_t)B * G * A * sizeof(_Float16);
    int useWmma = (ws_size >= needWmma && A <= MAX_A) ? 1 : 0;

    init_ws<<<(int)((nBA + 255) / 256), 256, 0, stream>>>(pack, normb, (int)nBA);

    if (useWmma) {
        dim3 gg((A + 127) / 128, (G + 15) / 16, B);
        gather_scores_wmma<<<gg, 256, 0, stream>>>(scores, gt_labels, g16, A, C, G);
    }

    yolo_match<<<B * G, 256, 0, stream>>>(scores, decode, anchors, gt_labels,
                                          gt_bboxes, gt_mask, g16, useWmma,
                                          pack, normb, A, C, G);

    float* out_bbox = (float*)d_out;                 // (B,A,4)
    float* out_cls  = out_bbox + nBA * 4;            // (B,A,C)
    float* out_fg   = out_cls + nBA * (size_t)C;     // (B,A)

    finalize_a<<<(int)((nBA + 255) / 256), 256, 0, stream>>>(
        pack, gt_bboxes, gt_labels, out_bbox, out_fg, clsw, A, G, (int)nBA);

    long long totCls = (long long)nBA * C;
    finalize_b<<<(unsigned)((totCls + 255) / 256), 256, 0, stream>>>(
        clsw, normb, out_cls, C, totCls);
}
